// TransformerGNNEdge_75522704933157
// MI455X (gfx1250) — matrix operand order repaired and checked by reading, not compiled, not run on hardware
//
#include <hip/hip_runtime.h>
#include <hip/hip_bf16.h>

// ---------------------------------------------------------------------------
// Edge-biased multi-head self attention (L=4096, D=256, H=8, DH=32, 14 edge
// types) for MI455X / gfx1250.
//
// Roofline: ~20 GFLOP vs 134 MB of int64 mask traffic (HBM floor ~6us @
// 23.3 TB/s) plus K/V re-reads from L2. Strategy:
//   * flash-attention (online softmax), never materialize (H,L,L) scores
//   * one workgroup = one 32-query tile, 8 waves = 8 heads -> each mask
//     element is fetched from HBM exactly once, shared via LDS
//   * 2 query sub-tiles per block so each K/V fragment fetch feeds 2x WMMAs
//     (halves L2-side K/V traffic vs 16-query blocks)
//   * mask tile streamed with GLOBAL_LOAD_ASYNC_TO_LDS_B64 (ASYNCcnt),
//     double-buffered so the next tile is in flight during WMMA/softmax
//   * all matmuls on v_wmma_f32_16x16x32_bf16 (f32 accumulate)
//   * weights pre-converted to bf16 once (not per tile-block)
// ---------------------------------------------------------------------------

#define L_  4096
#define D_  256
#define H_  8
#define DH_ 32
#define NE_ 14
#define TD3 (3 * D_)

typedef __attribute__((ext_vector_type(16))) __bf16         v16bf;
typedef __attribute__((ext_vector_type(8)))  float          v8f;
typedef __attribute__((ext_vector_type(8)))  unsigned short ushort8x; // 16 B
typedef __attribute__((ext_vector_type(2)))  int            v2i;

union BF16x16 {
    v16bf          v;
    unsigned short u[16];
    ushort8x       h[2];
};

// float -> bf16 bits, round-to-nearest-even (pure integer ops)
__device__ __forceinline__ unsigned short f2bf(float f) {
    unsigned u = __float_as_uint(f);
    unsigned r = (u + 0x7FFFu + ((u >> 16) & 1u)) >> 16;
    return (unsigned short)r;
}

__device__ __forceinline__ v8f wmma_bf16(const BF16x16& a, const BF16x16& b, v8f c) {
    return __builtin_amdgcn_wmma_f32_16x16x32_bf16(
        /*neg_a=*/false, a.v, /*neg_b=*/false, b.v,
        /*c_mod=*/(short)0, c, /*reuse_a=*/false, /*reuse_b=*/false);
}

// --------------------------- async LDS copy path ---------------------------
#if defined(__has_builtin)
#if __has_builtin(__builtin_amdgcn_global_load_async_to_lds_b64)
#define HAS_ASYNC_LDS 1
#endif
#endif

__device__ __forceinline__ void async_copy_b64(const long long* g, long long* l) {
#ifdef HAS_ASYNC_LDS
    // (v2i* [AS1 global], v2i* [AS3 lds], imm offset, imm cpol)
    __builtin_amdgcn_global_load_async_to_lds_b64(
        (__attribute__((address_space(1))) v2i*)(void*)g,
        (__attribute__((address_space(3))) v2i*)(void*)l,
        /*offset=*/0, /*cpol=*/0);
#else
    *l = *g;                            // sync fallback (DScnt, covered by barrier)
#endif
}

__device__ __forceinline__ void async_copy_wait() {
#ifdef HAS_ASYNC_LDS
#if __has_builtin(__builtin_amdgcn_s_wait_asynccnt)
    __builtin_amdgcn_s_wait_asynccnt(0);
#else
    asm volatile("s_wait_asynccnt 0" ::: "memory");
#endif
#endif
}

// ---------------------------------------------------------------------------
// Kernel 0: one-shot weight conversion f32 -> bf16 into workspace.
// ---------------------------------------------------------------------------
__global__ __launch_bounds__(256)
void cvt_weights_kernel(const float* __restrict__ in_w,    // (768,256)
                        const float* __restrict__ out_w,   // (256,256)
                        unsigned short* __restrict__ wq,   // 196608 bf16
                        unsigned short* __restrict__ wo) { // 65536 bf16
    const int idx = blockIdx.x * 256 + threadIdx.x;
    if (idx < TD3 * D_) {
        wq[idx] = f2bf(in_w[idx]);
    } else {
        const int j = idx - TD3 * D_;
        wo[j] = f2bf(out_w[j]);
    }
}

// ---------------------------------------------------------------------------
// Kernel 1: qkv = x @ in_proj_w^T + b   (4096x256) x (256x768)
// Writes Q,K as (H,L,DH) bf16 and V transposed (H,DH,L) bf16.
// Block: 256 threads = 8 waves, each wave one 16x16 tile. Grid: (L/16, 6).
// blockIdx.y -> output section is UNIFORM: {0,1}->Q {2,3}->K {4,5}->V.
// ---------------------------------------------------------------------------
__global__ __launch_bounds__(256)
void qkv_proj_kernel(const float* __restrict__ x,
                     const unsigned short* __restrict__ wbf, // (768,256) bf16
                     const float* __restrict__ b,            // (768,)
                     unsigned short* __restrict__ Qh,
                     unsigned short* __restrict__ Kh,
                     unsigned short* __restrict__ Vt) {
    constexpr int XS = 264;   // padded row stride: conflict-free ds_load_b128
    __shared__ unsigned short lx[16 * XS];

    const int m0 = blockIdx.x * 16;
    const int t  = threadIdx.x;

    for (int i = 0; i < 16; ++i) {
        int idx = i * 256 + t;
        int row = idx >> 8, col = idx & 255;
        lx[row * XS + col] = f2bf(x[(m0 + row) * D_ + col]);
    }
    __syncthreads();

    const int wid  = t >> 5;
    const int lane = t & 31;
    const int half = lane >> 4;
    const int r    = lane & 15;
    const int n    = blockIdx.y * 128 + wid * 16 + r;   // output column

    v8f acc = {};
    for (int kc = 0; kc < 8; ++kc) {
        const int kb = kc * 32;
        BF16x16 A, B;
        A.h[0] = *(const ushort8x*)&lx[r * XS + kb + half * 16];
        A.h[1] = *(const ushort8x*)&lx[r * XS + kb + half * 16 + 8];
        const ushort8x* wp = (const ushort8x*)&wbf[n * D_ + kb + half * 16];
        B.h[0] = wp[0];
        B.h[1] = wp[1];
        acc = wmma_bf16(A, B, acc);
    }

    const float bn   = b[n];
    const int   d    = n & 255;
    const int   head = d >> 5;
    const int   dh   = d & 31;
    const int   sec  = blockIdx.y >> 1;   // uniform scalar branch
    if (sec == 2) {
        for (int j = 0; j < 8; ++j) {
            const int m = m0 + j + half * 8;
            Vt[(head * DH_ + dh) * L_ + m] = f2bf(acc[j] + bn);
        }
    } else {
        unsigned short* __restrict__ dst = (sec == 0) ? Qh : Kh;
        for (int j = 0; j < 8; ++j) {
            const int m = m0 + j + half * 8;
            dst[(head * L_ + m) * DH_ + dh] = f2bf(acc[j] + bn);
        }
    }
}

// ---------------------------------------------------------------------------
// Kernel 2: flash attention with edge bias.
// Grid: L/32 blocks. Block: 256 threads = 8 waves; wave h = head h; each wave
// carries TWO 16-query sub-tiles so every K/V fragment fetch feeds 2x WMMAs.
// Mask tile (32x32 int64 = 8 KB) double-buffered in LDS via async copies.
// ---------------------------------------------------------------------------
__global__ __launch_bounds__(256)
void edge_attn_kernel(const unsigned short* __restrict__ Qh,
                      const unsigned short* __restrict__ Kh,
                      const unsigned short* __restrict__ Vt,
                      const long long* __restrict__ mask,   // (L,L) int64
                      const float* __restrict__ edge_bias,  // (14,8)
                      float* __restrict__ Ofull) {          // (L,D) f32
    constexpr int PS = 40;  // padded P row stride (80 B -> conflict-free b128)
    __shared__ long long      lmask[2][32 * 32];            // 2 x 8 KB
    __shared__ float          leb[NE_ * H_];
    __shared__ unsigned short lp[8 * 2 * 16 * PS];          // P staging, 2 q-tiles

    const int t = threadIdx.x;
    if (t < NE_ * H_) leb[t] = edge_bias[t];

    const int q0   = blockIdx.x * 32;
    const int wid  = t >> 5;
    const int lane = t & 31;
    const int half = lane >> 4;
    const int r    = lane & 15;
    const int h    = wid;

    // Q A-fragments for the two 16-query sub-tiles, reused across all keys.
    BF16x16 QA0, QA1;
    {
        const unsigned short* q0p = &Qh[(h * L_ + q0 + r) * DH_];
        QA0.h[0] = *(const ushort8x*)&q0p[half * 16];
        QA0.h[1] = *(const ushort8x*)&q0p[half * 16 + 8];
        const unsigned short* q1p = &Qh[(h * L_ + q0 + 16 + r) * DH_];
        QA1.h[0] = *(const ushort8x*)&q1p[half * 16];
        QA1.h[1] = *(const ushort8x*)&q1p[half * 16 + 8];
    }

    v8f   o00 = {}, o01 = {};        // q-tile 0, dh 0-15 / 16-31
    v8f   o10 = {}, o11 = {};        // q-tile 1
    float mr0[8], lr0[8], mr1[8], lr1[8];
    for (int j = 0; j < 8; ++j) {
        mr0[j] = -1e30f; lr0[j] = 0.0f;
        mr1[j] = -1e30f; lr1[j] = 0.0f;
    }
    const float scale = 0.17677669529663687f;   // 1/sqrt(32)

    unsigned short* myp0 = &lp[(wid * 2 + 0) * 16 * PS];
    unsigned short* myp1 = &lp[(wid * 2 + 1) * 16 * PS];

    // Prefetch mask tile 0 (32x32 int64 -> 4 x b64 per thread, coalesced).
    for (int u = 0; u < 4; ++u) {
        const int i   = u * 256 + t;
        const int row = i >> 5, col = i & 31;
        async_copy_b64(&mask[(long long)(q0 + row) * L_ + col], &lmask[0][i]);
    }

    for (int kt = 0; kt < 128; ++kt) {
        async_copy_wait();       // own async loads for tile kt done
        __syncthreads();         // everyone's done; prior tile's readers drained

        if (kt + 1 < 128) {      // prefetch tile kt+1 into the other buffer
            const int kn = (kt + 1) * 32;
            long long* buf = lmask[(kt + 1) & 1];
            for (int u = 0; u < 4; ++u) {
                const int i   = u * 256 + t;
                const int row = i >> 5, col = i & 31;
                async_copy_b64(&mask[(long long)(q0 + row) * L_ + kn + col], &buf[i]);
            }
        }

        const long long* cm = lmask[kt & 1];
        const int k0 = kt * 32;

        // S = Q K^T: K B-fragment loaded once, shared by both q-tiles.
        v8f s00, s01, s10, s11;     // s<qtile><keytile>
        {
            BF16x16 KB;
            const unsigned short* kr = &Kh[(h * L_ + k0 + r) * DH_ + half * 16];
            KB.h[0] = *(const ushort8x*)&kr[0];
            KB.h[1] = *(const ushort8x*)&kr[8];
            v8f z = {};
            s00 = wmma_bf16(QA0, KB, z);
            s10 = wmma_bf16(QA1, KB, z);
        }
        {
            BF16x16 KB;
            const unsigned short* kr = &Kh[(h * L_ + k0 + 16 + r) * DH_ + half * 16];
            KB.h[0] = *(const ushort8x*)&kr[0];
            KB.h[1] = *(const ushort8x*)&kr[8];
            v8f z = {};
            s01 = wmma_bf16(QA0, KB, z);
            s11 = wmma_bf16(QA1, KB, z);
        }

        // Bias + online softmax per q-tile. C/D layout: rows j+8*half,
        // column = r; width-16 shfl_xor reduces over keys within a row.
        for (int j = 0; j < 8; ++j) {
            const int row = j + half * 8;
            {   // q-tile 0: mask rows 0..15
                const int e0 = (int)cm[row * 32 + r];
                const int e1 = (int)cm[row * 32 + 16 + r];
                float a0 = s00[j] * scale + leb[e0 * H_ + h];
                float a1 = s01[j] * scale + leb[e1 * H_ + h];
                float bm = fmaxf(a0, a1);
                for (int dd = 8; dd >= 1; dd >>= 1) bm = fmaxf(bm, __shfl_xor(bm, dd, 16));
                const float mnew  = fmaxf(mr0[j], bm);
                const float alpha = __expf(mr0[j] - mnew);
                const float p0    = __expf(a0 - mnew);
                const float p1    = __expf(a1 - mnew);
                float ps = p0 + p1;
                for (int dd = 8; dd >= 1; dd >>= 1) ps += __shfl_xor(ps, dd, 16);
                lr0[j] = lr0[j] * alpha + ps;
                mr0[j] = mnew;
                o00[j] *= alpha;
                o01[j] *= alpha;
                myp0[row * PS + r]      = f2bf(p0);
                myp0[row * PS + 16 + r] = f2bf(p1);
            }
            {   // q-tile 1: mask rows 16..31
                const int e0 = (int)cm[(16 + row) * 32 + r];
                const int e1 = (int)cm[(16 + row) * 32 + 16 + r];
                float a0 = s10[j] * scale + leb[e0 * H_ + h];
                float a1 = s11[j] * scale + leb[e1 * H_ + h];
                float bm = fmaxf(a0, a1);
                for (int dd = 8; dd >= 1; dd >>= 1) bm = fmaxf(bm, __shfl_xor(bm, dd, 16));
                const float mnew  = fmaxf(mr1[j], bm);
                const float alpha = __expf(mr1[j] - mnew);
                const float p0    = __expf(a0 - mnew);
                const float p1    = __expf(a1 - mnew);
                float ps = p0 + p1;
                for (int dd = 8; dd >= 1; dd >>= 1) ps += __shfl_xor(ps, dd, 16);
                lr1[j] = lr1[j] * alpha + ps;
                mr1[j] = mnew;
                o10[j] *= alpha;
                o11[j] *= alpha;
                myp1[row * PS + r]      = f2bf(p0);
                myp1[row * PS + 16 + r] = f2bf(p1);
            }
        }

        // O += P V over K=32 keys; V B-fragments loaded once, used by both
        // q-tiles (V^T layout -> contiguous 32 B loads).
        BF16x16 VB0, VB1, PA0, PA1;
        const unsigned short* v0p = &Vt[(h * DH_ + r) * L_ + k0 + half * 16];
        VB0.h[0] = *(const ushort8x*)&v0p[0];
        VB0.h[1] = *(const ushort8x*)&v0p[8];
        const unsigned short* v1p = &Vt[(h * DH_ + 16 + r) * L_ + k0 + half * 16];
        VB1.h[0] = *(const ushort8x*)&v1p[0];
        VB1.h[1] = *(const ushort8x*)&v1p[8];
        PA0.h[0] = *(const ushort8x*)&myp0[r * PS + half * 16];
        PA0.h[1] = *(const ushort8x*)&myp0[r * PS + half * 16 + 8];
        PA1.h[0] = *(const ushort8x*)&myp1[r * PS + half * 16];
        PA1.h[1] = *(const ushort8x*)&myp1[r * PS + half * 16 + 8];
        o00 = wmma_bf16(PA0, VB0, o00);
        o01 = wmma_bf16(PA0, VB1, o01);
        o10 = wmma_bf16(PA1, VB0, o10);
        o11 = wmma_bf16(PA1, VB1, o11);
    }

    // Normalize and write merged-head output (L,D) f32.
    for (int j = 0; j < 8; ++j) {
        const int row = j + half * 8;
        const float i0 = 1.0f / lr0[j];
        const float i1 = 1.0f / lr1[j];
        Ofull[(q0 + row) * D_ + h * DH_ + r]           = o00[j] * i0;
        Ofull[(q0 + row) * D_ + h * DH_ + 16 + r]      = o01[j] * i0;
        Ofull[(q0 + 16 + row) * D_ + h * DH_ + r]      = o10[j] * i1;
        Ofull[(q0 + 16 + row) * D_ + h * DH_ + 16 + r] = o11[j] * i1;
    }
}

// ---------------------------------------------------------------------------
// Kernel 3: out = O @ out_proj_w^T + b   (4096x256) x (256x256) -> f32
// ---------------------------------------------------------------------------
__global__ __launch_bounds__(256)
void out_proj_kernel(const float* __restrict__ O,
                     const unsigned short* __restrict__ wbf, // (256,256) bf16
                     const float* __restrict__ b,
                     float* __restrict__ out) {
    constexpr int XS = 264;
    __shared__ unsigned short lx[16 * XS];

    const int m0 = blockIdx.x * 16;
    const int t  = threadIdx.x;

    for (int i = 0; i < 16; ++i) {
        int idx = i * 256 + t;
        int row = idx >> 8, col = idx & 255;
        lx[row * XS + col] = f2bf(O[(m0 + row) * D_ + col]);
    }
    __syncthreads();

    const int wid  = t >> 5;
    const int lane = t & 31;
    const int half = lane >> 4;
    const int r    = lane & 15;
    const int n    = blockIdx.y * 128 + wid * 16 + r;

    v8f acc = {};
    for (int kc = 0; kc < 8; ++kc) {
        const int kb = kc * 32;
        BF16x16 A, B;
        A.h[0] = *(const ushort8x*)&lx[r * XS + kb + half * 16];
        A.h[1] = *(const ushort8x*)&lx[r * XS + kb + half * 16 + 8];
        const ushort8x* wp = (const ushort8x*)&wbf[n * D_ + kb + half * 16];
        B.h[0] = wp[0];
        B.h[1] = wp[1];
        acc = wmma_bf16(A, B, acc);
    }

    const float bn = b[n];
    for (int j = 0; j < 8; ++j) {
        const int m = m0 + j + half * 8;
        out[m * D_ + n] = acc[j] + bn;
    }
}

// ---------------------------------------------------------------------------
extern "C" void kernel_launch(void* const* d_in, const int* in_sizes, int n_in,
                              void* d_out, int out_size, void* d_ws, size_t ws_size,
                              hipStream_t stream) {
    const float*     x      = (const float*)d_in[0];
    const long long* mask   = (const long long*)d_in[1];   // int64
    const float*     in_w   = (const float*)d_in[2];       // (768,256)
    const float*     in_b   = (const float*)d_in[3];       // (768,)
    const float*     out_w  = (const float*)d_in[4];       // (256,256)
    const float*     out_b  = (const float*)d_in[5];       // (256,)
    const float*     ebias  = (const float*)d_in[6];       // (14,8)
    (void)out_w;

    char* ws = (char*)d_ws;
    unsigned short* Qh    = (unsigned short*)(ws);                            // 2 MB
    unsigned short* Kh    = (unsigned short*)(ws + (size_t) 2 * 1024 * 1024); // 2 MB
    unsigned short* Vt    = (unsigned short*)(ws + (size_t) 4 * 1024 * 1024); // 2 MB
    float*          Ofull = (float*)         (ws + (size_t) 6 * 1024 * 1024); // 4 MB
    unsigned short* Wqbf  = (unsigned short*)(ws + (size_t)10 * 1024 * 1024); // 384 KB
    unsigned short* Wobf  = (unsigned short*)(ws + (size_t)10 * 1024 * 1024 + 512 * 1024);

    cvt_weights_kernel<<<dim3((TD3 * D_ + D_ * D_) / 256), 256, 0, stream>>>(
        in_w, out_w, Wqbf, Wobf);
    qkv_proj_kernel<<<dim3(L_ / 16, TD3 / 128), 256, 0, stream>>>(
        x, Wqbf, in_b, Qh, Kh, Vt);
    edge_attn_kernel<<<dim3(L_ / 32), 256, 0, stream>>>(
        Qh, Kh, Vt, mask, ebias, Ofull);
    out_proj_kernel<<<dim3(L_ / 16, D_ / 128), 256, 0, stream>>>(
        Ofull, Wobf, out_b, (float*)d_out);
}
